// CoreFinder_47459388621023
// MI455X (gfx1250) — compile-verified, matrix operand order checked by reading, not run on hardware
//
#include <hip/hip_runtime.h>
#include <cstdint>
#include <cstddef>

#define NVARS 50000
#define NCL   210000
#define NEDGE (NCL*3)
#define NGR   64
#define ROUNDS 16
#ifndef USE_ASYNC_LDS
#define USE_ASYNC_LDS 1
#endif

typedef __attribute__((ext_vector_type(16))) __bf16 v16bf;
typedef __attribute__((ext_vector_type(8)))  __bf16 v8bf;
typedef __attribute__((ext_vector_type(8)))  float  v8f;

union Frag { v16bf v; v8bf h[2]; };

__device__ __forceinline__ unsigned short f2bf(float f){
  unsigned u = __float_as_uint(f);
  unsigned r = u + 0x7fffu + ((u>>16)&1u);
  return (unsigned short)(r>>16);
}
__device__ __forceinline__ float sigm_(float x){ return 1.f/(1.f+expf(-x)); }
__device__ __forceinline__ float sp_(float x){ return x>15.f ? x : log1pf(expf(x)); }
__device__ __forceinline__ unsigned hash_u(unsigned x){
  x ^= x>>17; x*=0xed5ad4bbu; x^=x>>11; x*=0xac4c1b51u; x^=x>>15; x*=0x31848babu; x^=x>>14; return x;
}
__device__ __forceinline__ float gauss_(unsigned a, unsigned b, unsigned c){
  unsigned s1 = hash_u(a*0x9E3779B9u ^ b*0x85EBCA6Bu ^ (c+1u)*0xC2B2AE35u);
  unsigned s2 = hash_u(s1 ^ 0x27d4eb2fu);
  float u1 = ((s1>>8)+1u)*(1.f/16777217.f);
  float u2 = (s2>>8)*(1.f/16777216.f);
  return sqrtf(-2.f*logf(u1))*cosf(6.28318530718f*u2);
}

// ---------------- WMMA GEMM:  Y[M,N] = act(X[M,K] @ W[K,N] + b) ----------------
// Wt: bf16, K-transposed [N, Kp] (Kp = K padded to multiple of 32, zero-filled).
// Block: 128 threads = 4 wave32 waves; block tile = 64 rows x N cols.
__global__ void k_gemm_bf16(const float* __restrict__ X, const unsigned short* __restrict__ Wt,
                            const float* __restrict__ bias, float* __restrict__ Y,
                            int M, int K, int Kp, int N, int act)
{
  extern __shared__ unsigned short smem[];
  unsigned short* alds = smem;            // 64 x Kp  (activations, bf16)
  unsigned short* blds = smem + 64*Kp;    // N  x Kp  (weights^T, bf16)
  const int tid = threadIdx.x;
  const long rowBlk = (long)blockIdx.x * 64;

  // Stage A: f32 -> bf16 into LDS (zero-pad rows >= M and k >= K)
  for (int i = tid; i < 64*Kp; i += blockDim.x) {
    int r = i / Kp, k = i - r*Kp;
    long row = rowBlk + r;
    float v = (row < M && k < K) ? X[row*(long)K + k] : 0.f;
    alds[i] = f2bf(v);
  }
  // Stage B: bf16 weights via async global->LDS DMA (CDNA5 path)
#if USE_ASYNC_LDS
  {
    int nchunks = (N*Kp) >> 3;              // 16-byte chunks
    for (int c = tid; c < nchunks; c += blockDim.x) {
      unsigned ldsb = (unsigned)(uintptr_t)(&blds[c*8]);
      const unsigned short* g = Wt + (size_t)c*8;
      asm volatile("global_load_async_to_lds_b128 %0, %1, off"
                   :: "v"(ldsb), "v"(g) : "memory");
    }
    asm volatile("s_wait_asynccnt 0" ::: "memory");
  }
#else
  for (int i = tid; i < N*Kp; i += blockDim.x) blds[i] = Wt[i];
#endif
  __syncthreads();

  const int lane = tid & 31;
  const int wv   = tid >> 5;      // wave in block: 0..3
  const int hf   = lane >> 4;     // half-wave
  const int ml   = lane & 15;
  const int aRow = wv*16 + ml;    // A fragment row within block tile

  for (int nt = 0; nt < (N >> 4); ++nt) {
    float bc = bias[nt*16 + ml];  // bias depends only on output column
    v8f acc;
    #pragma unroll
    for (int r = 0; r < 8; ++r) acc[r] = bc;
    for (int kt = 0; kt < Kp; kt += 32) {
      Frag A, B;
      const int ka = aRow*Kp + kt + hf*8;
      A.h[0] = *(const v8bf*)(alds + ka);
      A.h[1] = *(const v8bf*)(alds + ka + 16);
      const int kb = (nt*16 + ml)*Kp + kt + hf*8;
      B.h[0] = *(const v8bf*)(blds + kb);
      B.h[1] = *(const v8bf*)(blds + kb + 16);
      acc = __builtin_amdgcn_wmma_f32_16x16x32_bf16(false, A.v, false, B.v,
                                                    (short)0, acc, false, false);
    }
    #pragma unroll
    for (int r = 0; r < 8; ++r) {
      long row = rowBlk + wv*16 + r + hf*8;
      float v = acc[r];
      if (act == 1) v = v > 0.f ? v : 0.2f*v;   // leaky_relu 0.2
      if (row < M) Y[row*(long)N + nt*16 + ml] = v;
    }
  }
}

// ---------------- small helper kernels ----------------
__global__ void k_fill(float* p, long n, float v){
  long i = (long)blockIdx.x*blockDim.x + threadIdx.x; if (i < n) p[i] = v;
}
__global__ void k_wt_bf16(const float* __restrict__ W, unsigned short* __restrict__ Wt,
                          int K, int N, int Kp){
  long i = (long)blockIdx.x*blockDim.x + threadIdx.x;
  if (i >= (long)N*Kp) return;
  int n = (int)(i / Kp), k = (int)(i - (long)n*Kp);
  float v = (k < K) ? W[(long)k*N + n] : 0.f;
  Wt[i] = f2bf(v);
}
__global__ void k_edges_deg(const int* __restrict__ lit, const int* __restrict__ cid,
                            const float* __restrict__ msig, const float* __restrict__ msp,
                            float* __restrict__ vsum, float* __restrict__ ldeg){
  long e = (long)blockIdx.x*blockDim.x + threadIdx.x;
  if (e >= NEDGE) return;
  int c = cid[e], l = lit[e];
  atomicAdd(&vsum[l], msp[c]);
  atomicAdd(&ldeg[l], msig[c]);
}
__global__ void k_var_pre(const float* vsum, const float* ldeg, const int* vgid,
                          float* vg_val, float* vdw, float* vg_sum){
  long v = (long)blockIdx.x*blockDim.x + threadIdx.x;
  if (v >= NVARS) return;
  float s = vsum[v] + vsum[v + NVARS];
  float g = 1.f - expf(-s);
  vg_val[v] = g;
  float ld = ldeg[v] + ldeg[v + NVARS];
  vdw[v] = 4.f * rsqrtf(fmaxf(ld, 1.f));
  atomicAdd(&vg_sum[vgid[v]], g);
}
__global__ void k_dwlit(const float* ldeg, float* dwl){
  long l = (long)blockIdx.x*blockDim.x + threadIdx.x;
  if (l >= 2L*NVARS) return;
  dwl[l] = rsqrtf(fmaxf(ldeg[l], 1.f));
}
__global__ void k_cg_sum(const float* msig, const int* cgid, float* cg_sum){
  long c = (long)blockIdx.x*blockDim.x + threadIdx.x;
  if (c >= NCL) return;
  atomicAdd(&cg_sum[cgid[c]], msig[c]);
}
__global__ void k_vgnorm(const float* vg_val, const int* vgid, const float* vg_sum, float* vg_norm){
  long v = (long)blockIdx.x*blockDim.x + threadIdx.x;
  if (v >= NVARS) return;
  float s = vg_sum[vgid[v]];
  vg_norm[v] = vg_val[v] * (s > 0.f ? 1.f/s : 0.f);
}
__global__ void k_cgnorm(const float* msig, const int* cgid, const float* cg_sum, float* cg_norm){
  long c = (long)blockIdx.x*blockDim.x + threadIdx.x;
  if (c >= NCL) return;
  float s = cg_sum[cgid[c]];
  cg_norm[c] = msig[c] * (s > 0.f ? 1.f/s : 0.f);
}
__global__ void k_build_xq(const float* __restrict__ vars, float* __restrict__ xq, int rnd){
  long i = (long)blockIdx.x*blockDim.x + threadIdx.x;
  if (i >= (long)NVARS*68) return;
  int v = (int)(i / 68), j = (int)(i - (long)v*68);
  xq[i] = (j < 64) ? vars[(long)v*64 + j] : gauss_((unsigned)rnd, (unsigned)v, (unsigned)(j-64));
}
__global__ void k_lits(const float* __restrict__ q, float* __restrict__ lits){
  long i = (long)blockIdx.x*blockDim.x + threadIdx.x;
  if (i >= (long)NVARS*64) return;
  float x = q[i];
  lits[i] = sp_(x);
  lits[i + (long)NVARS*64] = sp_(-x);
}
__global__ void k_cv(const float* __restrict__ lits, const int* __restrict__ lit,
                     const float* __restrict__ msig, float* __restrict__ cv){
  long i = (long)blockIdx.x*blockDim.x + threadIdx.x;
  if (i >= (long)NCL*64) return;
  int c = (int)(i >> 6), f = (int)(i & 63);
  long e = (long)c*3;
  float s = lits[(long)lit[e]*64+f] + lits[(long)lit[e+1]*64+f] + lits[(long)lit[e+2]*64+f];
  cv[i] = expf(-msig[c]*s);
}
__global__ void k_scatter_edges(const float* __restrict__ src, int lds,
                                const int* __restrict__ lit, const int* __restrict__ cid,
                                const float* __restrict__ msig, float* __restrict__ dst){
  long i = (long)blockIdx.x*blockDim.x + threadIdx.x;
  if (i >= (long)NEDGE*64) return;
  long e = i >> 6; int f = (int)(i & 63);
  int c = cid[e];
  atomicAdd(&dst[(long)lit[e]*64 + f], msig[c]*src[(long)c*lds + f]);
}
__global__ void k_unit_grad(const float* __restrict__ q, const float* __restrict__ g2,
                            const float* __restrict__ vdw, const float* __restrict__ vars,
                            float* __restrict__ unit){
  long i = (long)blockIdx.x*blockDim.x + threadIdx.x;
  if (i >= (long)NVARS*64) return;
  int v = (int)(i >> 6), f = (int)(i & 63);
  float qq = q[i];
  float g = (-g2[i]*sigm_(qq) + g2[i + (long)NVARS*64]*sigm_(-qq)) * vdw[v];
  unit[(long)v*256 + f]      = g;
  unit[(long)v*256 + 64 + f] = vars[i];
}
__global__ void k_unit_vl(const float* __restrict__ vl, const float* __restrict__ dwl,
                          float* __restrict__ unit){
  long i = (long)blockIdx.x*blockDim.x + threadIdx.x;
  if (i >= (long)NVARS*64) return;
  int v = (int)(i >> 6), f = (int)(i & 63);
  unit[(long)v*256 + 128 + f] = vl[i] * dwl[v];
  unit[(long)v*256 + 192 + f] = vl[i + (long)NVARS*64] * dwl[v + NVARS];
}
__global__ void k_cunit(const float* __restrict__ cstate, const float* __restrict__ cv,
                        float* __restrict__ cunit){
  long i = (long)blockIdx.x*blockDim.x + threadIdx.x;
  if (i >= (long)NCL*64) return;
  long c = i >> 6; int f = (int)(i & 63);
  cunit[c*128 + f]      = cstate[i];
  cunit[c*128 + 64 + f] = 4.f*cv[i];
}
__global__ void k_pn_mean(const float* __restrict__ x, int ldx, const int* __restrict__ gid,
                          const float* __restrict__ gval, float* __restrict__ mean_g, long rows){
  long i = (long)blockIdx.x*blockDim.x + threadIdx.x;
  if (i >= rows*64) return;
  long r = i >> 6; int f = (int)(i & 63);
  atomicAdd(&mean_g[(long)gid[r]*64 + f], gval[r]*x[r*ldx + f]);
}
__global__ void k_pn_center(const float* __restrict__ x, int ldx, const int* __restrict__ gid,
                            const float* __restrict__ gval, const float* __restrict__ mean_g,
                            float* __restrict__ var_g, float* __restrict__ x2, long rows){
  long r = blockIdx.x; int f = threadIdx.x;
  __shared__ float red[64];
  if (r >= rows) return;
  int g = gid[r]; float gv = gval[r];
  float v = x[r*ldx + f] - gv*mean_g[(long)g*64 + f];
  x2[r*64 + f] = v;
  red[f] = v*v;
  __syncthreads();
  for (int s = 32; s > 0; s >>= 1){ if (f < s) red[f] += red[f+s]; __syncthreads(); }
  if (f == 0) atomicAdd(&var_g[g], gv*red[0]*(1.f/64.f));
}
__global__ void k_pn_final(const float* __restrict__ x2, const int* __restrict__ gid,
                           const float* __restrict__ gval, const float* __restrict__ var_g,
                           const float* __restrict__ gain, const float* __restrict__ bias,
                           float* __restrict__ state, long rows){
  long i = (long)blockIdx.x*blockDim.x + threadIdx.x;
  if (i >= rows*64) return;
  long r = i >> 6; int f = (int)(i & 63);
  float pn = x2[i]*rsqrtf(gval[r]*var_g[gid[r]] + 1e-6f)*gain[f] + bias[f];
  state[i] = 0.25f*pn + 0.1f*state[i];
}
__global__ void k_logits(const float* __restrict__ h, const float* __restrict__ w,
                         const float* __restrict__ b, float* __restrict__ logits){
  long v = (long)blockIdx.x*blockDim.x + threadIdx.x;
  if (v >= NVARS) return;
  float s = b[0];
  #pragma unroll 8
  for (int f = 0; f < 64; ++f) s += h[v*64 + f]*w[f];
  logits[v] = s;
}
__global__ void k_loss(const float* __restrict__ logits, const int* __restrict__ lit,
                       const float* __restrict__ msig, const int* __restrict__ cgid,
                       float* __restrict__ pg){
  long c = (long)blockIdx.x*blockDim.x + threadIdx.x;
  if (c >= NCL) return;
  float a = msig[c]; float s = 0.f;
  for (int j = 0; j < 3; ++j){
    int l = lit[c*3 + j];
    float lg = (l < NVARS) ? logits[l] : -logits[l - NVARS];
    s += a*sp_(lg);
  }
  float cvv = expf(-s);
  float mixed = cvv * -logf(1.f - cvv + 1e-8f);
  atomicAdd(&pg[cgid[c]], a*mixed);
}
__global__ void k_acc(const float* pg, float* accb){
  int g = threadIdx.x;
  if (g < NGR) accb[g] += sqrtf(pg[g] + 1e-6f) - 1.0e-3f;
}
__global__ void k_writeout(const float* logits, const float* accb, float* out){
  long i = (long)blockIdx.x*blockDim.x + threadIdx.x;
  if (i < NVARS) out[i] = logits[i];
  else if (i < NVARS + NGR) out[i] = accb[i - NVARS] * (1.f/ROUNDS);
}

static inline unsigned gblk(long n){ return (unsigned)((n + 255)/256); }

extern "C" void kernel_launch(void* const* d_in, const int* in_sizes, int n_in,
                              void* d_out, int out_size, void* d_ws, size_t ws_size,
                              hipStream_t stream) {
  (void)in_sizes; (void)n_in; (void)out_size; (void)ws_size;
  const int*   lit_idx    = (const int*)  d_in[0];
  const int*   clause_idx = (const int*)  d_in[1];
  const int*   clause_gid = (const int*)  d_in[2];
  const int*   var_gid    = (const int*)  d_in[3];
  const float* mask_sig   = (const float*)d_in[4];
  const float* mask_sp    = (const float*)d_in[5];
  const float* qW0 = (const float*)d_in[6];  const float* qb0 = (const float*)d_in[7];
  const float* qW1 = (const float*)d_in[8];  const float* qb1 = (const float*)d_in[9];
  const float* cW0 = (const float*)d_in[10]; const float* cb0 = (const float*)d_in[11];
  const float* cW1 = (const float*)d_in[12]; const float* cb1 = (const float*)d_in[13];
  const float* uW0 = (const float*)d_in[14]; const float* ub0 = (const float*)d_in[15];
  const float* uW1 = (const float*)d_in[16]; const float* ub1 = (const float*)d_in[17];
  const float* uW2 = (const float*)d_in[18]; const float* ub2 = (const float*)d_in[19];
  const float* oW0 = (const float*)d_in[20]; const float* ob0 = (const float*)d_in[21];
  const float* oW1 = (const float*)d_in[22]; const float* ob1 = (const float*)d_in[23];
  const float* vgain = (const float*)d_in[24]; const float* vbias = (const float*)d_in[25];
  const float* cgain = (const float*)d_in[26]; const float* cbias = (const float*)d_in[27];

  char* base = (char*)d_ws; size_t off = 0;
  auto alloc = [&](size_t bytes)->void*{
    off = (off + 255) & ~(size_t)255; void* p = base + off; off += bytes; return p;
  };
  float* variables = (float*)alloc((size_t)NVARS*64*4);
  float* cstate    = (float*)alloc((size_t)NCL*64*4);
  float* xq        = (float*)alloc((size_t)NVARS*68*4);
  float* qh        = (float*)alloc((size_t)NVARS*64*4);
  float* query     = (float*)alloc((size_t)NVARS*64*4);
  float* lits      = (float*)alloc((size_t)2*NVARS*64*4);
  float* cv        = (float*)alloc((size_t)NCL*64*4);
  float* g2        = (float*)alloc((size_t)2*NVARS*64*4);
  float* cunit     = (float*)alloc((size_t)NCL*128*4);
  float* chid      = (float*)alloc((size_t)NCL*128*4);
  float* cdata     = cunit;  // alias: cunit dead once chid computed
  float* x2c       = (float*)alloc((size_t)NCL*64*4);
  float* vl        = (float*)alloc((size_t)2*NVARS*64*4);
  float* unitb     = (float*)alloc((size_t)NVARS*256*4);
  float* uh1       = (float*)alloc((size_t)NVARS*128*4);
  float* uh2       = (float*)alloc((size_t)NVARS*128*4);
  float* newv      = (float*)alloc((size_t)NVARS*64*4);
  float* x2v       = (float*)alloc((size_t)NVARS*64*4);
  float* outh      = (float*)alloc((size_t)NVARS*64*4);
  float* logits    = (float*)alloc((size_t)NVARS*4);
  float* var_sum   = (float*)alloc((size_t)2*NVARS*4);
  float* lit_deg   = (float*)alloc((size_t)2*NVARS*4);
  float* vg_val    = (float*)alloc((size_t)NVARS*4);
  float* vdw       = (float*)alloc((size_t)NVARS*4);
  float* dwl       = (float*)alloc((size_t)2*NVARS*4);
  float* vg_norm   = (float*)alloc((size_t)NVARS*4);
  float* cg_norm   = (float*)alloc((size_t)NCL*4);
  float* vg_sum    = (float*)alloc(NGR*4);
  float* cg_sum    = (float*)alloc(NGR*4);
  float* mean_g    = (float*)alloc(NGR*64*4);
  float* var_g     = (float*)alloc(NGR*4);
  float* pg        = (float*)alloc(NGR*4);
  float* accb      = (float*)alloc(NGR*4);
  unsigned short* qW0t = (unsigned short*)alloc((size_t)64*96*2);
  unsigned short* qW1t = (unsigned short*)alloc((size_t)64*64*2);
  unsigned short* cW0t = (unsigned short*)alloc((size_t)128*128*2);
  unsigned short* cW1t = (unsigned short*)alloc((size_t)128*128*2);
  unsigned short* uW0t = (unsigned short*)alloc((size_t)128*256*2);
  unsigned short* uW1t = (unsigned short*)alloc((size_t)128*128*2);
  unsigned short* uW2t = (unsigned short*)alloc((size_t)64*128*2);
  unsigned short* oW0t = (unsigned short*)alloc((size_t)64*64*2);

  auto zero = [&](float* p, long n){ k_fill<<<gblk(n), 256, 0, stream>>>(p, n, 0.f); };
  auto gemm = [&](const float* X, const unsigned short* Wt, const float* b, float* Y,
                  long M, int K, int Kp, int N, int act){
    size_t sm = (size_t)(64 + N) * (size_t)Kp * 2;
    k_gemm_bf16<<<dim3((unsigned)((M + 63)/64)), dim3(128), sm, stream>>>(
        X, Wt, b, Y, (int)M, K, Kp, N, act);
  };

  // ---- one-time (per launch) precompute ----
  zero(var_sum, 2L*NVARS); zero(lit_deg, 2L*NVARS);
  zero(vg_sum, NGR); zero(cg_sum, NGR); zero(accb, NGR);
  k_fill<<<gblk((long)NVARS*64), 256, 0, stream>>>(variables, (long)NVARS*64, 1.f);
  k_fill<<<gblk((long)NCL*64),   256, 0, stream>>>(cstate,   (long)NCL*64,   1.f);
  k_edges_deg<<<gblk(NEDGE), 256, 0, stream>>>(lit_idx, clause_idx, mask_sig, mask_sp, var_sum, lit_deg);
  k_var_pre<<<gblk(NVARS), 256, 0, stream>>>(var_sum, lit_deg, var_gid, vg_val, vdw, vg_sum);
  k_dwlit<<<gblk(2L*NVARS), 256, 0, stream>>>(lit_deg, dwl);
  k_cg_sum<<<gblk(NCL), 256, 0, stream>>>(mask_sig, clause_gid, cg_sum);
  k_vgnorm<<<gblk(NVARS), 256, 0, stream>>>(vg_val, var_gid, vg_sum, vg_norm);
  k_cgnorm<<<gblk(NCL), 256, 0, stream>>>(mask_sig, clause_gid, cg_sum, cg_norm);
  k_wt_bf16<<<gblk(64L*96),   256, 0, stream>>>(qW0, qW0t,  68,  64,  96);
  k_wt_bf16<<<gblk(64L*64),   256, 0, stream>>>(qW1, qW1t,  64,  64,  64);
  k_wt_bf16<<<gblk(128L*128), 256, 0, stream>>>(cW0, cW0t, 128, 128, 128);
  k_wt_bf16<<<gblk(128L*128), 256, 0, stream>>>(cW1, cW1t, 128, 128, 128);
  k_wt_bf16<<<gblk(128L*256), 256, 0, stream>>>(uW0, uW0t, 256, 128, 256);
  k_wt_bf16<<<gblk(128L*128), 256, 0, stream>>>(uW1, uW1t, 128, 128, 128);
  k_wt_bf16<<<gblk(64L*128),  256, 0, stream>>>(uW2, uW2t, 128,  64, 128);
  k_wt_bf16<<<gblk(64L*64),   256, 0, stream>>>(oW0, oW0t,  64,  64,  64);

  // ---- 16 message-passing rounds ----
  for (int r = 0; r < ROUNDS; ++r) {
    k_build_xq<<<gblk((long)NVARS*68), 256, 0, stream>>>(variables, xq, r);
    gemm(xq, qW0t, qb0, qh,    NVARS, 68, 96, 64, 1);
    gemm(qh, qW1t, qb1, query, NVARS, 64, 64, 64, 0);
    k_lits<<<gblk((long)NVARS*64), 256, 0, stream>>>(query, lits);
    k_cv<<<gblk((long)NCL*64), 256, 0, stream>>>(lits, lit_idx, mask_sig, cv);
    zero(g2, 2L*NVARS*64);
    k_scatter_edges<<<gblk((long)NEDGE*64), 256, 0, stream>>>(cv, 64, lit_idx, clause_idx, mask_sig, g2);
    k_unit_grad<<<gblk((long)NVARS*64), 256, 0, stream>>>(query, g2, vdw, variables, unitb);
    k_cunit<<<gblk((long)NCL*64), 256, 0, stream>>>(cstate, cv, cunit);
    gemm(cunit, cW0t, cb0, chid,  NCL, 128, 128, 128, 1);
    gemm(chid,  cW1t, cb1, cdata, NCL, 128, 128, 128, 0);
    zero(vl, 2L*NVARS*64);
    k_scatter_edges<<<gblk((long)NEDGE*64), 256, 0, stream>>>(cdata, 128, lit_idx, clause_idx, mask_sig, vl);
    k_unit_vl<<<gblk((long)NVARS*64), 256, 0, stream>>>(vl, dwl, unitb);
    // clause pair-norm on cdata[:, 64:128] -> cstate
    zero(mean_g, (long)NGR*64); zero(var_g, NGR);
    k_pn_mean<<<gblk((long)NCL*64), 256, 0, stream>>>(cdata + 64, 128, clause_gid, cg_norm, mean_g, NCL);
    k_pn_center<<<dim3(NCL), dim3(64), 0, stream>>>(cdata + 64, 128, clause_gid, cg_norm, mean_g, var_g, x2c, NCL);
    k_pn_final<<<gblk((long)NCL*64), 256, 0, stream>>>(x2c, clause_gid, cg_norm, var_g, cgain, cbias, cstate, NCL);
    // update MLP
    gemm(unitb, uW0t, ub0, uh1,  NVARS, 256, 256, 128, 1);
    gemm(uh1,   uW1t, ub1, uh2,  NVARS, 128, 128, 128, 1);
    gemm(uh2,   uW2t, ub2, newv, NVARS, 128, 128,  64, 0);
    // variable pair-norm -> variables
    zero(mean_g, (long)NGR*64); zero(var_g, NGR);
    k_pn_mean<<<gblk((long)NVARS*64), 256, 0, stream>>>(newv, 64, var_gid, vg_norm, mean_g, NVARS);
    k_pn_center<<<dim3(NVARS), dim3(64), 0, stream>>>(newv, 64, var_gid, vg_norm, mean_g, var_g, x2v, NVARS);
    k_pn_final<<<gblk((long)NVARS*64), 256, 0, stream>>>(x2v, var_gid, vg_norm, var_g, vgain, vbias, variables, NVARS);
    // output head + per-graph loss
    gemm(variables, oW0t, ob0, outh, NVARS, 64, 64, 64, 1);
    k_logits<<<gblk(NVARS), 256, 0, stream>>>(outh, oW1, ob1, logits);
    zero(pg, NGR);
    k_loss<<<gblk(NCL), 256, 0, stream>>>(logits, lit_idx, mask_sig, clause_gid, pg);
    k_acc<<<dim3(1), dim3(64), 0, stream>>>(pg, accb);
  }
  k_writeout<<<gblk(NVARS + NGR), 256, 0, stream>>>(logits, accb, (float*)d_out);
}
